// MutilRegionAttention_25812753449021
// MI455X (gfx1250) — matrix operand order repaired and checked by reading, not compiled, not run on hardware
//
#include <hip/hip_runtime.h>
#include <stdint.h>

// ---------------------------------------------------------------------------
// MI455X (gfx1250) MultiRegionAttention
// B=4, C=64, H=W=256, K=8 -> K2=64, L=1024; R=16 -> N=256
// Dominant work: q = A Wq^T and v = A Wv^T with A = 64x1024 per (b,c),
// done as bf16 v_wmma_f32_16x16x32_bf16 with f32 accumulate,
// 2 N-tiles per wave for A-fragment reuse + independent WMMA chains.
// kmean factors through mean(A) -> tiny GEMM. softmax(q)·kmean is fused
// (online softmax) so q is never materialized. A panel staged into LDS with
// global_load_async_to_lds_b128 (ASYNCcnt path).
// ---------------------------------------------------------------------------

typedef __bf16 bf16;
typedef __attribute__((ext_vector_type(16))) __bf16 v16bf;
typedef __attribute__((ext_vector_type(8)))  __bf16 v8bf;
typedef __attribute__((ext_vector_type(8)))  float  v8f;

#define MB_ (1024ull * 1024ull)
// workspace byte offsets
#define WQ_B   (0ull)          // bf16 [1024*1024]   Wq   (N,K) row-major
#define WK_B   (2ull  * MB_)   // bf16 [1024*1024]   Wk   (= Wkv rows 0..1023)
#define WV_B   (4ull  * MB_)   // bf16 [1024*1024]   Wv   (= Wkv rows 1024..2047)
#define WCA_B  (6ull  * MB_)   // bf16 [64*64]       Wc[:, :64]  (o,ci)
#define XBF_B  (8ull  * MB_)   // bf16 [256*64*1024] packed A panels
#define MBF_B  (40ull * MB_)   // bf16 [256*1024]    column means of A
#define KM_B   (41ull * MB_)   // f32  [256*1024]    kmean
#define XR_B   (42ull * MB_)   // f32  [4*256*64]    region pooled x
#define ZR_B   (43ull * MB_)   // f32  [4*64*256]    region contribution to z
#define INS_B  (44ull * MB_)   // bf16 [4*256*256*64] inside, channel-last

__device__ __forceinline__ bf16 f2bf(float f) {
  union { float f; uint32_t u; } v; v.f = f;
  uint32_t r = v.u + 0x7FFFu + ((v.u >> 16) & 1u);   // round-to-nearest-even
  union { uint16_t s; bf16 b; } o; o.s = (uint16_t)(r >> 16);
  return o.b;
}

// A-operand fragment: contiguous 16 bf16 at p (k = 16*half + s, matches B)
__device__ __forceinline__ v16bf load_frag_a(const bf16* p) {
  const v8bf* q = (const v8bf*)p;
  v8bf lo = q[0], hi = q[1];
  v16bf r;
#pragma unroll
  for (int i = 0; i < 8; ++i) { r[i] = lo[i]; r[i + 8] = hi[i]; }
  return r;
}
// B-operand fragment: contiguous 16 bf16 (col-major B == row-major (N,K) weight)
__device__ __forceinline__ v16bf load_frag_b(const bf16* p) {
  const v8bf* q = (const v8bf*)p;
  v8bf lo = q[0], hi = q[1];
  v16bf r;
#pragma unroll
  for (int i = 0; i < 8; ++i) { r[i] = lo[i]; r[i + 8] = hi[i]; }
  return r;
}

// row reductions within each 16-lane half of the wave32 (D-tile rows live in
// one half; xor masks 1..8 never cross the half boundary)
__device__ __forceinline__ float hredmax(float x) {
#pragma unroll
  for (int m = 1; m < 16; m <<= 1) x = fmaxf(x, __shfl_xor(x, m, 32));
  return x;
}
__device__ __forceinline__ float hredsum(float x) {
#pragma unroll
  for (int m = 1; m < 16; m <<= 1) x += __shfl_xor(x, m, 32);
  return x;
}

// online softmax + Σexp·kmean update for one 16x16 q tile
__device__ __forceinline__ void online_update(const v8f& acc, float kmv,
                                              float* m8, float* s8, float* t8) {
#pragma unroll
  for (int r = 0; r < 8; ++r) {
    float qv = acc[r];
    float nm = fmaxf(m8[r], hredmax(qv));
    float sc = __expf(m8[r] - nm);
    float p  = __expf(qv - nm);
    float ps = hredsum(p);
    float pt = hredsum(p * kmv);
    s8[r] = s8[r] * sc + ps;
    t8[r] = t8[r] * sc + pt;
    m8[r] = nm;
  }
}

// scatter one scaled v tile to channel-last inside buffer
__device__ __forceinline__ void store_inside(bf16* ins, const v8f& acc,
                                             const float* sig8, int b, int c,
                                             int mBase, int hlf, int l) {
  int hp = l >> 5, wp = l & 31;
#pragma unroll
  for (int r = 0; r < 8; ++r) {
    int k2 = mBase + r + 8 * hlf;
    int kh = k2 >> 3, kw = k2 & 7;
    int h = hp * 8 + kh, w = wp * 8 + kw;
    ins[(((size_t)b * 256 + h) * 256 + w) * 64 + c] = f2bf(acc[r] * sig8[r]);
  }
}

// ---------------------------------------------------------------------------
__global__ void k_convert(const float* __restrict__ Wq,
                          const float* __restrict__ Wkv,
                          const float* __restrict__ Wc, char* ws) {
  bf16* wq  = (bf16*)(ws + WQ_B);
  bf16* wk  = (bf16*)(ws + WK_B);
  bf16* wv  = (bf16*)(ws + WV_B);
  bf16* wca = (bf16*)(ws + WCA_B);
  const int total = 3 * 1048576 + 4096;
  for (int i = blockIdx.x * blockDim.x + threadIdx.x; i < total;
       i += gridDim.x * blockDim.x) {
    if (i < 1048576)            wq[i]             = f2bf(Wq[i]);
    else if (i < 2 * 1048576)   wk[i - 1048576]   = f2bf(Wkv[i - 1048576]);
    else if (i < 3 * 1048576)   wv[i - 2*1048576] = f2bf(Wkv[i - 1048576]);
    else {
      int j = i - 3 * 1048576; int o = j >> 6, ci = j & 63;
      wca[j] = f2bf(Wc[o * 128 + ci]);
    }
  }
}

// pack x -> bf16 A panels (b,c,k2,l) + column means over k2
__global__ void k_pack(const float* __restrict__ x, char* ws) {
  int id = blockIdx.x * blockDim.x + threadIdx.x;   // B*C*L = 262144
  bf16* xbf = (bf16*)(ws + XBF_B);
  bf16* mbf = (bf16*)(ws + MBF_B);
  int bc = id >> 10, l = id & 1023;
  int hp = l >> 5, wp = l & 31;
  const float* src = x + (size_t)bc * 65536 + hp * 2048 + wp * 8;
  bf16* dst = xbf + (size_t)bc * 65536 + l;
  float sum = 0.f;
#pragma unroll
  for (int kh = 0; kh < 8; ++kh)
#pragma unroll
    for (int kw = 0; kw < 8; ++kw) {
      float v = src[kh * 256 + kw];
      sum += v;
      dst[(kh * 8 + kw) * 1024] = f2bf(v);
    }
  mbf[bc * 1024 + l] = f2bf(sum * (1.f / 64.f));
}

// 16x16 region pooling: xr[b,n,c]
__global__ void k_pool(const float* __restrict__ x, char* ws) {
  int id = blockIdx.x * blockDim.x + threadIdx.x;   // 65536
  float* xr = (float*)(ws + XR_B);
  int b = id >> 14, n = (id >> 6) & 255, c = id & 63;
  int hr = n >> 4, wr = n & 15;
  const float* src = x + (size_t)(b * 64 + c) * 65536 + hr * 16 * 256 + wr * 16;
  float s = 0.f;
  for (int i = 0; i < 16; ++i)
#pragma unroll
    for (int j = 0; j < 16; ++j) s += src[i * 256 + j];
  xr[(b * 256 + n) * 64 + c] = s * (1.f / 256.f);
}

// kmean = mean(A) · Wk^T : M=256, N=1024, K=1024, bf16 WMMA
__global__ void __launch_bounds__(128) k_kmean(char* ws) {
  const bf16* mbf = (const bf16*)(ws + MBF_B);
  const bf16* wk  = (const bf16*)(ws + WK_B);
  float* km = (float*)(ws + KM_B);
  int wave = threadIdx.x >> 5, lane = threadIdx.x & 31;
  int hlf = lane >> 4, lr = lane & 15;
  int mt = blockIdx.x >> 4, ng = blockIdx.x & 15;
  int mBase = mt * 16, nBase = (ng * 4 + wave) * 16;
  const bf16* Ap = mbf + (mBase + lr) * 1024 + hlf * 16;
  const bf16* Bp = wk + (size_t)(nBase + lr) * 1024 + hlf * 16;
  v8f acc = {};
  for (int k = 0; k < 1024; k += 32) {
    v16bf a  = load_frag_a(Ap + k);
    v16bf bb = load_frag_b(Bp + k);
    acc = __builtin_amdgcn_wmma_f32_16x16x32_bf16(false, a, false, bb,
                                                  (short)0, acc, false, false);
  }
#pragma unroll
  for (int r = 0; r < 8; ++r)
    km[(size_t)(mBase + r + 8 * hlf) * 1024 + nBase + lr] = acc[r];
}

// region attention (tiny): one block per batch, thread n handles region n.
// Also folds the region half of the 1x1 conv into zr[b,o,n].
__global__ void k_region(const float* __restrict__ Wqr,
                         const float* __restrict__ Wkvr,
                         const float* __restrict__ Wc, char* ws) {
  int b = blockIdx.x, n = threadIdx.x;
  const float* xr = (const float*)(ws + XR_B);
  float* zr = (float*)(ws + ZR_B);
  __shared__ float krm[64];
  if (n < 64) krm[n] = 0.f;
  __syncthreads();
  float xrow[64];
#pragma unroll 8
  for (int c = 0; c < 64; ++c) xrow[c] = xr[(b * 256 + n) * 64 + c];
  for (int j = 0; j < 64; ++j) {
    float s = 0.f;
    for (int c = 0; c < 64; ++c) s += xrow[c] * Wkvr[j * 64 + c];
    atomicAdd(&krm[j], s * (1.f / 256.f));
  }
  __syncthreads();
  float qr[64]; float mx = -1e30f;
  for (int j = 0; j < 64; ++j) {
    float s = 0.f;
    for (int c = 0; c < 64; ++c) s += xrow[c] * Wqr[j * 64 + c];
    qr[j] = s; mx = fmaxf(mx, s);
  }
  float S = 0.f, T = 0.f;
  for (int j = 0; j < 64; ++j) {
    float p = __expf(qr[j] - mx);
    S += p; T += p * krm[j];
  }
  float av = T / S;
  float sig = 1.f / (1.f + __expf(-av));
  float rvv[64];
  for (int j = 0; j < 64; ++j) {
    float s = 0.f;
    for (int c = 0; c < 64; ++c) s += xrow[c] * Wkvr[(64 + j) * 64 + c];
    rvv[j] = sig * s;
  }
  for (int o = 0; o < 64; ++o) {
    float s = 0.f;
    for (int j = 0; j < 64; ++j) s += rvv[j] * Wc[o * 128 + 64 + j];
    zr[(size_t)(b * 64 + o) * 256 + n] = s;
  }
}

// Main fused kernel: one WG per (b,c). A panel (64x1024 bf16) resident in LDS,
// staged via global_load_async_to_lds_b128 (ASYNCcnt).
// Phase 1: q tiles via WMMA (2 N-tiles/wave) + online softmax stats.
// Phase 2: v tiles via WMMA, scaled by sigmoid(a), channel-last bf16 store.
__global__ void __launch_bounds__(128) k_inside(char* ws) {
  extern __shared__ char smem_raw[];
  bf16* Apanel = (bf16*)smem_raw;                    // 64*1024 bf16 = 128 KB
  int bc = blockIdx.x;
  int b = bc >> 6, c = bc & 63;
  const bf16* xbf = (const bf16*)(ws + XBF_B);
  const bf16* wq  = (const bf16*)(ws + WQ_B);
  const bf16* wv  = (const bf16*)(ws + WV_B);
  const float* km = (const float*)(ws + KM_B);
  bf16* ins = (bf16*)(ws + INS_B);

  {  // async 128 KB A-panel copy: global -> LDS, no VGPR staging (ASYNCcnt)
    const char* gsrc = (const char*)(xbf + (size_t)bc * 65536);
    unsigned lbase = (unsigned)(size_t)Apanel;       // LDS byte offset
    for (int i = threadIdx.x; i < 8192; i += 128) {
      unsigned long long src = (unsigned long long)(gsrc + (size_t)i * 16);
      unsigned dst = lbase + (unsigned)i * 16u;
      asm volatile("global_load_async_to_lds_b128 %0, %1, off"
                   :: "v"(dst), "v"(src) : "memory");
    }
    asm volatile("s_wait_asynccnt 0" ::: "memory");
  }
  __syncthreads();

  int wave = threadIdx.x >> 5, lane = threadIdx.x & 31;
  int hlf = lane >> 4, lr = lane & 15;
  int mBase = wave * 16;
  const bf16* Arow = Apanel + (mBase + lr) * 1024 + hlf * 16;

  float m8[8], s8[8], t8[8];
#pragma unroll
  for (int r = 0; r < 8; ++r) { m8[r] = -1e30f; s8[r] = 0.f; t8[r] = 0.f; }

  // -------- phase 1: q, fused online softmax·kmean (2 tiles per pass) ------
  for (int nt = 0; nt < 64; nt += 2) {
    const bf16* Bp0 = wq + (size_t)(nt * 16 + lr) * 1024 + hlf * 16;
    const bf16* Bp1 = Bp0 + 16 * 1024;
    v8f acc0 = {}, acc1 = {};
    for (int k = 0; k < 1024; k += 32) {
      __builtin_prefetch(Bp0 + k + 512, 0, 0);
      v16bf a  = load_frag_a(Arow + k);
      v16bf b0 = load_frag_b(Bp0 + k);
      v16bf b1 = load_frag_b(Bp1 + k);
      acc0 = __builtin_amdgcn_wmma_f32_16x16x32_bf16(false, a, false, b0,
                                                     (short)0, acc0, false, false);
      acc1 = __builtin_amdgcn_wmma_f32_16x16x32_bf16(false, a, false, b1,
                                                     (short)0, acc1, false, false);
    }
    float kmv0 = km[(size_t)bc * 1024 + nt * 16 + lr];
    float kmv1 = km[(size_t)bc * 1024 + nt * 16 + 16 + lr];
    online_update(acc0, kmv0, m8, s8, t8);
    online_update(acc1, kmv1, m8, s8, t8);
  }
  float sig8[8];
#pragma unroll
  for (int r = 0; r < 8; ++r) {
    float av = t8[r] / s8[r];
    sig8[r] = 1.f / (1.f + __expf(-av));
  }

  // -------- phase 2: v, scale by sigmoid(a), store channel-last ------------
  for (int nt = 0; nt < 64; nt += 2) {
    const bf16* Bp0 = wv + (size_t)(nt * 16 + lr) * 1024 + hlf * 16;
    const bf16* Bp1 = Bp0 + 16 * 1024;
    v8f acc0 = {}, acc1 = {};
    for (int k = 0; k < 1024; k += 32) {
      __builtin_prefetch(Bp0 + k + 512, 0, 0);
      v16bf a  = load_frag_a(Arow + k);
      v16bf b0 = load_frag_b(Bp0 + k);
      v16bf b1 = load_frag_b(Bp1 + k);
      acc0 = __builtin_amdgcn_wmma_f32_16x16x32_bf16(false, a, false, b0,
                                                     (short)0, acc0, false, false);
      acc1 = __builtin_amdgcn_wmma_f32_16x16x32_bf16(false, a, false, b1,
                                                     (short)0, acc1, false, false);
    }
    store_inside(ins, acc0, sig8, b, c, mBase, hlf, nt * 16 + lr);
    store_inside(ins, acc1, sig8, b, c, mBase, hlf, nt * 16 + 16 + lr);
  }
}

// Epilogue: 1x1 conv over inside (WMMA, M=pixels N=64 K=64) + region bias +
// BN + sigmoid, times x. One wave = 16 pixels (one h row, 16 consecutive w).
__global__ void __launch_bounds__(128) k_epilogue(
    const float* __restrict__ x, const float* __restrict__ bcv,
    const float* __restrict__ gamma, const float* __restrict__ beta,
    const float* __restrict__ rmv, const float* __restrict__ rvv,
    char* ws, float* __restrict__ out) {
  const bf16* ins  = (const bf16*)(ws + INS_B);
  const bf16* wca  = (const bf16*)(ws + WCA_B);
  const float* zr  = (const float*)(ws + ZR_B);
  int id = blockIdx.x;                // B*H*4 = 4096
  int b = id >> 10;
  int h = (id >> 2) & 255;
  int wq4 = id & 3;
  int wave = threadIdx.x >> 5, lane = threadIdx.x & 31;
  int hlf = lane >> 4, lr = lane & 15;
  int wBase = wq4 * 64 + wave * 16;
  int region = (h >> 4) * 16 + (wBase >> 4);
  const bf16* Ap = ins + (((size_t)b * 256 + h) * 256 + wBase + lr) * 64 + hlf * 16;
#pragma unroll
  for (int nt = 0; nt < 4; ++nt) {
    int o = nt * 16 + lr;
    const bf16* Bp = wca + o * 64 + hlf * 16;
    v8f acc = {};
#pragma unroll
    for (int k = 0; k < 64; k += 32) {
      v16bf a  = load_frag_a(Ap + k);
      v16bf bb = load_frag_b(Bp + k);
      acc = __builtin_amdgcn_wmma_f32_16x16x32_bf16(false, a, false, bb,
                                                    (short)0, acc, false, false);
    }
    float sc = gamma[o] * rsqrtf(rvv[o] + 1e-5f);
    float sh = beta[o] - rmv[o] * sc;
    float zb = zr[((size_t)b * 64 + o) * 256 + region] + bcv[o];
    const float* xp = x   + ((size_t)(b * 64 + o) * 256 + h) * 256;
    float*       op = out + ((size_t)(b * 64 + o) * 256 + h) * 256;
#pragma unroll
    for (int r = 0; r < 8; ++r) {
      int w = wBase + r + 8 * hlf;
      float z = (acc[r] + zb) * sc + sh;
      float sg = 1.f / (1.f + __expf(-z));
      op[w] = sg * xp[w];
    }
  }
}

// ---------------------------------------------------------------------------
extern "C" void kernel_launch(void* const* d_in, const int* in_sizes, int n_in,
                              void* d_out, int out_size, void* d_ws,
                              size_t ws_size, hipStream_t stream) {
  (void)in_sizes; (void)n_in; (void)out_size; (void)ws_size;
  const float* x     = (const float*)d_in[0];
  const float* Wq    = (const float*)d_in[1];
  const float* Wkv   = (const float*)d_in[2];
  const float* Wqr   = (const float*)d_in[3];
  const float* Wkvr  = (const float*)d_in[4];
  const float* Wc    = (const float*)d_in[5];
  const float* bcv   = (const float*)d_in[6];
  const float* gamma = (const float*)d_in[7];
  const float* beta  = (const float*)d_in[8];
  const float* rmv   = (const float*)d_in[9];
  const float* rvv   = (const float*)d_in[10];
  char* ws = (char*)d_ws;
  float* out = (float*)d_out;

  k_convert <<<2048, 256, 0, stream>>>(Wq, Wkv, Wc, ws);
  k_pack    <<<1024, 256, 0, stream>>>(x, ws);
  k_pool    <<<256,  256, 0, stream>>>(x, ws);
  k_kmean   <<<256,  128, 0, stream>>>(ws);
  k_region  <<<4,    256, 0, stream>>>(Wqr, Wkvr, Wc, ws);
  k_inside  <<<256,  128, 131072, stream>>>(ws);
  k_epilogue<<<4096, 128, 0, stream>>>(x, bcv, gamma, beta, rmv, rvv, ws, out);
}
